// EncdecMultiheadAttn_41996190220759
// MI455X (gfx1250) — matrix operand order repaired and checked by reading, not compiled, not run on hardware
//
#include <hip/hip_runtime.h>

// ---------------------------------------------------------------------------
// Types for CDNA5 WMMA (wave32): A/B = 16 bf16 per lane, C/D = 8 f32 per lane
// ---------------------------------------------------------------------------
typedef __bf16 bf16_t;
typedef __bf16 bf16x16 __attribute__((ext_vector_type(16)));
typedef float  f32x8   __attribute__((ext_vector_type(8)));
typedef int    i32x4   __attribute__((ext_vector_type(4)));

union FragU { bf16x16 v; i32x4 q[2]; };

__device__ __forceinline__ f32x8 zero_f32x8() {
    f32x8 z = {0.f, 0.f, 0.f, 0.f, 0.f, 0.f, 0.f, 0.f};
    return z;
}

// A-matrix fragment (16x32 bf16, row-major src, leading dim ld):
// lane<16: row=lane,    K = {0..7} and {16..23}
// lane>=16: row=lane-16, K = {8..15} and {24..31}
__device__ __forceinline__ bf16x16 load_fragA(const bf16_t* tile, int ld) {
    const int lane = threadIdx.x & 31;
    const bf16_t* p = tile + (size_t)(lane & 15) * ld + ((lane >> 4) << 3);
    FragU u;
    u.q[0] = *(const i32x4*)(p);
    u.q[1] = *(const i32x4*)(p + 16);
    return u.v;
}

// B-matrix fragment (32x16 bf16) built from a row-major matrix W[N][K]:
// B[k][n] = W[n][k]. lane<16: col n=lane, K={0..7,16..23}; lane>=16: col n=lane-16, K={8..15,24..31}.
// Per-lane data = two runs of 8 bf16 (16 apart) of one W row.
__device__ __forceinline__ bf16x16 load_fragB(const bf16_t* tile, int ld) {
    const int lane = threadIdx.x & 31;
    const bf16_t* p = tile + (size_t)(lane & 15) * ld + ((lane >> 4) << 3);
    FragU u;
    u.q[0] = *(const i32x4*)(p);
    u.q[1] = *(const i32x4*)(p + 16);
    return u.v;
}

__device__ __forceinline__ f32x8 wmma_bf16(bf16x16 a, bf16x16 b, f32x8 c) {
    return __builtin_amdgcn_wmma_f32_16x16x32_bf16(
        /*neg_a=*/false, a, /*neg_b=*/false, b,
        /*c_mod=*/(short)0, c, /*reuse_a=*/false, /*reuse_b=*/false);
}

__device__ __forceinline__ unsigned pack_bf16(float lo, float hi) {
    union { bf16_t h[2]; unsigned u; } pk;
    pk.h[0] = (bf16_t)lo;
    pk.h[1] = (bf16_t)hi;
    return pk.u;
}

// Swap values between lane halves (lane <-> lane^16) in VALU via permlanex16;
// falls back to shfl (ds_bpermute) if the builtin is unavailable.
__device__ __forceinline__ float half_swap(float v) {
#if __has_builtin(__builtin_amdgcn_permlanex16)
    return __int_as_float(__builtin_amdgcn_permlanex16(
        __float_as_int(v), __float_as_int(v), 0x76543210, 0xfedcba98, false, false));
#else
    return __shfl_xor(v, 16, 32);
#endif
}

// ---------------------------------------------------------------------------
// Problem constants: E=1024, H=16, D=64, Bc=4, Tq=Tk=2048, M = Tq*Bc = 8192
// ---------------------------------------------------------------------------

// ---------------------------------------------------------------------------
// fp32 -> bf16 conversion (vectorized by 4)
// ---------------------------------------------------------------------------
__global__ __launch_bounds__(256)
void cvt_f32_to_bf16(const float* __restrict__ x, bf16_t* __restrict__ y, int n4) {
    int i = blockIdx.x * blockDim.x + threadIdx.x;
    if (i < n4) {
        float4 v = ((const float4*)x)[i];
        union { bf16_t h[4]; unsigned long long u; } pk;
        pk.h[0] = (bf16_t)v.x; pk.h[1] = (bf16_t)v.y;
        pk.h[2] = (bf16_t)v.z; pk.h[3] = (bf16_t)v.w;
        ((unsigned long long*)y)[i] = pk.u;
    }
}

// ---------------------------------------------------------------------------
// LayerNorm over last dim (1024) -> bf16. One block (256 thr) per row.
// ---------------------------------------------------------------------------
__global__ __launch_bounds__(256)
void layernorm_bf16(const float* __restrict__ x, const float* __restrict__ gamma,
                    const float* __restrict__ beta, bf16_t* __restrict__ y) {
    const int row = blockIdx.x;
    const float4 v = ((const float4*)(x + (size_t)row * 1024))[threadIdx.x];
    float s  = v.x + v.y + v.z + v.w;
    float s2 = v.x * v.x + v.y * v.y + v.z * v.z + v.w * v.w;
#pragma unroll
    for (int m = 1; m < 32; m <<= 1) {
        s  += __shfl_xor(s,  m, 32);
        s2 += __shfl_xor(s2, m, 32);
    }
    __shared__ float sh[16];
    const int wave = threadIdx.x >> 5;
    if ((threadIdx.x & 31) == 0) { sh[wave] = s; sh[8 + wave] = s2; }
    __syncthreads();
    s = 0.f; s2 = 0.f;
#pragma unroll
    for (int w = 0; w < 8; w++) { s += sh[w]; s2 += sh[8 + w]; }
    const float mu   = s * (1.0f / 1024.0f);
    const float var  = s2 * (1.0f / 1024.0f) - mu * mu;
    const float rstd = rsqrtf(var + 1e-5f);
    const float4 g = ((const float4*)gamma)[threadIdx.x];
    const float4 b = ((const float4*)beta)[threadIdx.x];
    union { bf16_t h[4]; unsigned long long u; } pk;
    pk.h[0] = (bf16_t)((v.x - mu) * rstd * g.x + b.x);
    pk.h[1] = (bf16_t)((v.y - mu) * rstd * g.y + b.y);
    pk.h[2] = (bf16_t)((v.z - mu) * rstd * g.z + b.z);
    pk.h[3] = (bf16_t)((v.w - mu) * rstd * g.w + b.w);
    *(unsigned long long*)(y + (size_t)row * 1024 + threadIdx.x * 4) = pk.u;
}

// ---------------------------------------------------------------------------
// WMMA GEMM: C[m,n] = sum_k A[m,k] * W[n,k]     (A: MxK bf16, W: NxK bf16)
// Block = 256 thr (8 waves as 2x4), wave tile 64x64, block tile 128x256.
// 16 WMMA per 16 b128 loads per k-step (vs 8/12 with 32x64).
// MODE 0: q-proj  -> out0[b][h][t][d] = 0.125*C           (bf16, head-major)
// MODE 1: kv-proj -> out0 = K[b][h][s][d], out1 = V^T[b][h][d][s]  (bf16)
// MODE 2: out-proj-> outF[m,n] = resid[m,n] + C           (fp32 + residual)
// ---------------------------------------------------------------------------
template <int MODE>
__global__ __launch_bounds__(256)
void gemm_bf16_wmma(const bf16_t* __restrict__ A, const bf16_t* __restrict__ W,
                    int M, int N, int K,
                    bf16_t* __restrict__ out0, bf16_t* __restrict__ out1,
                    const float* __restrict__ resid, float* __restrict__ outF) {
    const int wave = threadIdx.x >> 5;
    const int lane = threadIdx.x & 31;
    const int m0 = blockIdx.y * 128 + (wave >> 2) * 64;
    const int n0 = blockIdx.x * 256 + (wave & 3) * 64;

    f32x8 acc[4][4];
#pragma unroll
    for (int i = 0; i < 4; i++)
#pragma unroll
        for (int j = 0; j < 4; j++) acc[i][j] = zero_f32x8();

    for (int k0 = 0; k0 < K; k0 += 32) {
        // prefetch this lane's A/W stream 1KB (16 k-tiles) ahead
        __builtin_prefetch(A + (size_t)(m0 + (lane & 15)) * K + k0 + 512, 0, 1);
        __builtin_prefetch(W + (size_t)(n0 + (lane & 15)) * K + k0 + 512, 0, 1);
        bf16x16 a[4], b[4];
#pragma unroll
        for (int i = 0; i < 4; i++)
            a[i] = load_fragA(A + (size_t)(m0 + i * 16) * K + k0, K);
#pragma unroll
        for (int j = 0; j < 4; j++)
            b[j] = load_fragB(W + (size_t)(n0 + j * 16) * K + k0, K);
#pragma unroll
        for (int i = 0; i < 4; i++)
#pragma unroll
            for (int j = 0; j < 4; j++)
                acc[i][j] = wmma_bf16(a[i], b[j], acc[i][j]);
    }

    // C layout: lane n=lane&15, hi=lane>>4; reg r -> element (row r+8*hi, col n)
    const int nlane = lane & 15;
    const int hi = lane >> 4;
#pragma unroll
    for (int i = 0; i < 4; i++) {
#pragma unroll
        for (int j = 0; j < 4; j++) {
#pragma unroll
            for (int r = 0; r < 8; r++) {
                const int m = m0 + i * 16 + r + hi * 8;
                const int f = n0 + j * 16 + nlane;
                const float v = acc[i][j][r];
                if constexpr (MODE == 0) {
                    const int t = m >> 2, bb = m & 3;      // rows are (t,b), B=4
                    const int h = f >> 6, d = f & 63;      // f = h*64 + d
                    out0[((size_t)((bb * 16 + h) * 2048 + t)) * 64 + d] =
                        (bf16_t)(v * 0.125f);              // * D^-0.5
                } else if constexpr (MODE == 1) {
                    const int s = m >> 2, bb = m & 3;
                    const int h = f >> 7, rem = f & 127;   // f = h*128 + {k:0..63, v:64..127}
                    if (rem < 64)
                        out0[((size_t)((bb * 16 + h) * 2048 + s)) * 64 + rem] = (bf16_t)v;
                    else
                        out1[((size_t)((bb * 16 + h) * 64 + (rem - 64))) * 2048 + s] = (bf16_t)v;
                } else {
                    const size_t idx = (size_t)m * N + f;
                    outF[idx] = resid[idx] + v;
                }
            }
        }
    }
}

// ---------------------------------------------------------------------------
// Flash attention per (b,h), fully transposed formulation:
//   S^T = K . Q^T  (so each lane owns one q-column t -> in-lane softmax)
//   O^T += V^T . P^T
// Block = 256 thr (8 waves), each wave owns 16 q columns; grid (Tq/128, B*H).
// q,k head-major [t|s][d]; v stored transposed [d][s].
// ---------------------------------------------------------------------------
__global__ __launch_bounds__(256)
void attn_flash_wmma(const bf16_t* __restrict__ qb, const bf16_t* __restrict__ kb,
                     const bf16_t* __restrict__ vtb, bf16_t* __restrict__ ctx) {
    __shared__ bf16_t lds_buf[8][16 * 64];  // per wave: P^T tile (1KB) / out tile (2KB)
    const int wave = threadIdx.x >> 5;
    const int lane = threadIdx.x & 31;
    const int bh = blockIdx.y;             // b*16 + h
    const int b = bh >> 4, h = bh & 15;
    const int t0 = blockIdx.x * 128 + wave * 16;
    const int col = lane & 15;             // this lane's q column (t = t0+col)
    const int hi = lane >> 4;

    const bf16_t* qh = qb  + (size_t)bh * 2048 * 64;   // [t][d]
    const bf16_t* kh = kb  + (size_t)bh * 2048 * 64;   // [s][d]
    const bf16_t* vh = vtb + (size_t)bh * 64 * 2048;   // [d][s]
    bf16_t* myp = &lds_buf[wave][0];                   // P^T as [t][s], ld=32

    // Q as B-fragments (lane col = t), d = 0..31 and 32..63
    const bf16x16 q0 = load_fragB(qh + (size_t)t0 * 64,      64);
    const bf16x16 q1 = load_fragB(qh + (size_t)t0 * 64 + 32, 64);

    f32x8 o[4];                            // O^T: rows d = j*16 + r + 8*hi, col t
#pragma unroll
    for (int j = 0; j < 4; j++) o[j] = zero_f32x8();
    float mrun = -3.0e38f, lrun = 0.f;     // per-lane (per q column)

    for (int s0 = 0; s0 < 2048; s0 += 32) {
        // S^T tiles: c0 rows s0..s0+15, c1 rows s0+16..s0+31; cols = q columns
        f32x8 c0 = zero_f32x8(), c1 = zero_f32x8();
        c0 = wmma_bf16(load_fragA(kh + (size_t)s0 * 64,            64), q0, c0);
        c0 = wmma_bf16(load_fragA(kh + (size_t)s0 * 64 + 32,       64), q1, c0);
        c1 = wmma_bf16(load_fragA(kh + (size_t)(s0 + 16) * 64,      64), q0, c1);
        c1 = wmma_bf16(load_fragA(kh + (size_t)(s0 + 16) * 64 + 32, 64), q1, c1);

        // in-lane max over this lane's 16 s-values, then combine partner half
        float mx = fmaxf(c0[0], c1[0]);
#pragma unroll
        for (int r = 1; r < 8; r++) mx = fmaxf(mx, fmaxf(c0[r], c1[r]));
        mx = fmaxf(mx, half_swap(mx));
        const float nm = fmaxf(mrun, mx);
        const float sc = __expf(mrun - nm);
        mrun = nm;

        float p0[8], p1[8], ps = 0.f;
#pragma unroll
        for (int r = 0; r < 8; r++) {
            p0[r] = __expf(c0[r] - nm);
            p1[r] = __expf(c1[r] - nm);
            ps += p0[r] + p1[r];
        }
        ps += half_swap(ps);
        lrun = lrun * sc + ps;
#pragma unroll
        for (int j = 0; j < 4; j++) o[j] *= sc;

        // store P^T to LDS as [t][s] (ld=32), bf16 pairs packed into b32
        // this lane holds (s = r+8*hi) from c0 and (s = r+8*hi+16) from c1
        unsigned* myp32 = (unsigned*)myp;
#pragma unroll
        for (int r = 0; r < 8; r += 2) {
            const int e = col * 32 + hi * 8 + r;            // even element index
            myp32[e >> 1]        = pack_bf16(p0[r], p0[r + 1]);
            myp32[(e + 16) >> 1] = pack_bf16(p1[r], p1[r + 1]);
        }
        // P^T as one B-fragment (shared by all 4 d-tiles); DS in-order per wave
        const bf16x16 pB = load_fragB(myp, 32);
#pragma unroll
        for (int j = 0; j < 4; j++) {
            const bf16x16 vA = load_fragA(vh + (size_t)(j * 16) * 2048 + s0, 2048);
            o[j] = wmma_bf16(vA, pB, o[j]);
        }
    }

    // normalize and stage O^T -> LDS as [t][d] (ld=64), bf16 pairs
    const float inv = 1.0f / lrun;
    unsigned* mo32 = (unsigned*)myp;
#pragma unroll
    for (int j = 0; j < 4; j++) {
#pragma unroll
        for (int r = 0; r < 8; r += 2) {
            const int e = col * 64 + j * 16 + hi * 8 + r;   // even element index
            mo32[e >> 1] = pack_bf16(o[j][r] * inv, o[j][r + 1] * inv);
        }
    }
    // coalesced store: each row of 16 is 64 bf16 = 32 lanes x b32
#pragma unroll 4
    for (int row = 0; row < 16; row++) {
        const unsigned val = mo32[row * 32 + lane];
        *(unsigned*)(ctx + (size_t)((t0 + row) * 4 + b) * 1024 + h * 64 + lane * 2) = val;
    }
}

// ---------------------------------------------------------------------------
// Host-side orchestration
// ---------------------------------------------------------------------------
extern "C" void kernel_launch(void* const* d_in, const int* in_sizes, int n_in,
                              void* d_out, int out_size, void* d_ws, size_t ws_size,
                              hipStream_t stream) {
    (void)in_sizes; (void)n_in; (void)out_size; (void)ws_size;
    const float* query = (const float*)d_in[0];   // (2048, 4, 1024)
    const float* key   = (const float*)d_in[1];   // (2048, 4, 1024)
    const float* gamma = (const float*)d_in[2];   // (1024,)
    const float* beta  = (const float*)d_in[3];   // (1024,)
    const float* w_q   = (const float*)d_in[4];   // (1024, 1024)
    const float* w_kv  = (const float*)d_in[5];   // (2048, 1024)
    const float* w_out = (const float*)d_in[6];   // (1024, 1024)
    float* out = (float*)d_out;

    // Workspace layout (bytes)
    char* ws = (char*)d_ws;
    const size_t SZ_ACT = (size_t)8192 * 1024 * 2;   // 16 MiB (bf16 8192x1024)
    const size_t SZ_WQ  = (size_t)1024 * 1024 * 2;   //  2 MiB
    const size_t SZ_WKV = (size_t)2048 * 1024 * 2;   //  4 MiB
    size_t off = 0;
    bf16_t* lnb   = (bf16_t*)(ws + off); off += SZ_ACT;   // LN(query) bf16
    bf16_t* keyb  = (bf16_t*)(ws + off); off += SZ_ACT;   // key bf16
    bf16_t* wqb   = (bf16_t*)(ws + off); off += SZ_WQ;
    bf16_t* wkvb  = (bf16_t*)(ws + off); off += SZ_WKV;
    bf16_t* woutb = (bf16_t*)(ws + off); off += SZ_WQ;
    bf16_t* qbuf  = (bf16_t*)(ws + off); off += SZ_ACT;   // [b][h][t][d]
    bf16_t* kbuf  = (bf16_t*)(ws + off); off += SZ_ACT;   // [b][h][s][d]
    bf16_t* vtbuf = (bf16_t*)(ws + off); off += SZ_ACT;   // [b][h][d][s]
    bf16_t* ctxb  = (bf16_t*)(ws + off); off += SZ_ACT;   // (t,b,e) bf16

    // 1) conversions
    cvt_f32_to_bf16<<<(8388608 / 4) / 256, 256, 0, stream>>>(key,   keyb,  8388608 / 4);
    cvt_f32_to_bf16<<<(1048576 / 4) / 256, 256, 0, stream>>>(w_q,   wqb,   1048576 / 4);
    cvt_f32_to_bf16<<<(2097152 / 4) / 256, 256, 0, stream>>>(w_kv,  wkvb,  2097152 / 4);
    cvt_f32_to_bf16<<<(1048576 / 4) / 256, 256, 0, stream>>>(w_out, woutb, 1048576 / 4);

    // 2) LayerNorm(query) -> bf16
    layernorm_bf16<<<8192, 256, 0, stream>>>(query, gamma, beta, lnb);

    // 3) q = LN(query) @ w_q^T * D^-0.5  -> head-major
    gemm_bf16_wmma<0><<<dim3(1024 / 256, 8192 / 128), 256, 0, stream>>>(
        lnb, wqb, 8192, 1024, 1024, qbuf, nullptr, nullptr, nullptr);

    // 4) kv = key @ w_kv^T -> K head-major, V transposed head-major
    gemm_bf16_wmma<1><<<dim3(2048 / 256, 8192 / 128), 256, 0, stream>>>(
        keyb, wkvb, 8192, 2048, 1024, kbuf, vtbuf, nullptr, nullptr);

    // 5) flash attention per (b,h), transposed formulation
    attn_flash_wmma<<<dim3(2048 / 128, 64), 256, 0, stream>>>(qbuf, kbuf, vtbuf, ctxb);

    // 6) out = query + ctx @ w_out^T
    gemm_bf16_wmma<2><<<dim3(1024 / 256, 8192 / 128), 256, 0, stream>>>(
        ctxb, woutb, 8192, 1024, 1024, nullptr, nullptr, query, out);
}
